// BuildingEnergyForecastingModel_27221502722587
// MI455X (gfx1250) — matrix operand order repaired and checked by reading, not compiled, not run on hardware
//
#include <hip/hip_runtime.h>
#include <hip/hip_bf16.h>
#include <math.h>

// ---------------------------------------------------------------------------
// Types for CDNA5 WMMA (gfx1250, wave32)
// ---------------------------------------------------------------------------
typedef __bf16 bf16_t;
typedef __attribute__((ext_vector_type(16))) __bf16 v16bf;
typedef __attribute__((ext_vector_type(2)))  __bf16 bf16x2;
typedef __attribute__((ext_vector_type(8)))  float  v8f;

#define D_MODEL 1024
#define SEQ_N   1024
#define BATCH   8
#define NHEAD   16
#define HEAD_DIM 64
#define MOE_HID 2048
#define NEXP    4
#define TOKENS  (BATCH * SEQ_N)   // 8192

#define LDS_PITCH 40   // bf16 elements per staged row (80 B: 16B-aligned, bank-conflict-free)

__device__ inline v8f wmma_bf16(v16bf a, v16bf b, v8f c) {
  // D = A(16x32) * B(32x16) + C(16x16), f32 accumulate
  return __builtin_amdgcn_wmma_f32_16x16x32_bf16(false, a, false, b, (short)0, c,
                                                 false, false);
}

// A-operand fragment: 16x32 bf16 tile, row-major, contiguous k order (k = 16*(lane/16) + s), same as the B loaders.
// lane 0-15: M = lane, K = 0..15; lane 16-31: M = lane-16, K = 16..31
__device__ inline v16bf load_frag_A(const bf16_t* __restrict__ A, int lda, int lane) {
  int m  = lane & 15;
  int kh = (lane >> 4) << 4;        // 0 or 16
  const bf16_t* row = A + (size_t)m * lda;
  v16bf f;
#pragma unroll
  for (int j = 0; j < 8; ++j) {
    int k = kh + (j << 1);
    bf16x2 p = *reinterpret_cast<const bf16x2*>(row + k);
    f[2 * j]     = p[0];
    f[2 * j + 1] = p[1];
  }
  return f;
}

// B-operand (32x16) built from 16 rows of a row-major matrix W[N,K]:
// B[k][n] = W[n][k]  (i.e. computes A * W^T).  Contiguous k order: lane n (0-15) holds
// K=0..15 of column n across 8 VGPRs (2 packed per VGPR); lanes 16-31 hold K=16..31.
__device__ inline v16bf load_frag_B_rows(const bf16_t* __restrict__ W, int ldw, int lane) {
  int n  = lane & 15;
  int kg = (lane >> 4) << 4;        // 0 or 16
  const bf16_t* row = W + (size_t)n * ldw + kg;
  v16bf f;
#pragma unroll
  for (int j = 0; j < 8; ++j) {
    bf16x2 p = *reinterpret_cast<const bf16x2*>(row + 2 * j);
    f[2 * j]     = p[0];
    f[2 * j + 1] = p[1];
  }
  return f;
}

// B-operand (32x16) from a column slice of a row-major matrix V[K, N]:
// B[k][n] = V[k][n]   (strided loads down columns)
__device__ inline v16bf load_frag_B_cols(const bf16_t* __restrict__ V, int ldv, int lane) {
  int n  = lane & 15;
  int kg = (lane >> 4) << 4;
  v16bf f;
#pragma unroll
  for (int j = 0; j < 8; ++j) {
    f[2 * j]     = V[(size_t)(kg + 2 * j)     * ldv + n];
    f[2 * j + 1] = V[(size_t)(kg + 2 * j + 1) * ldv + n];
  }
  return f;
}

// ---------------------------------------------------------------------------
// CDNA5 async-DMA staging: global -> LDS, ASYNCcnt-tracked (inline asm since
// the clang builtin arity differs across toolchains; asm is portable).
// ---------------------------------------------------------------------------
__device__ inline void async_ld_b128(unsigned lds_off, const void* gaddr) {
  asm volatile("global_load_async_to_lds_b128 %0, %1, off"
               :: "v"(lds_off), "v"(gaddr) : "memory");
}
__device__ inline void wait_async0() {
  asm volatile("s_wait_asynccnt 0x0" ::: "memory");
}

// Stage a 128 x 32 bf16 tile (row stride ldg elements, starting col k0) into
// LDS tile lds[128][LDS_PITCH].  512 x b128 chunks; 256 threads x 2 chunks.
__device__ inline void stage_tile_async(const bf16_t* __restrict__ gbase, int ldg,
                                        bf16_t (*lds)[LDS_PITCH], int k0) {
  int t = threadIdx.x;
#pragma unroll
  for (int i = 0; i < 2; ++i) {
    int c = t + i * 256;
    int row = c >> 2;
    int seg = c & 3;               // 16-byte segment within the 64-byte row
    const void* g = (const void*)(gbase + (size_t)row * ldg + k0 + seg * 8);
    unsigned l = (unsigned)(uintptr_t)&lds[row][seg * 8];
    async_ld_b128(l, g);
  }
}

// ---------------------------------------------------------------------------
// bf16 WMMA GEMM:  C[M,N] = act(A[M,K] * W[N,K]^T + bias)
// Block tile 128(M) x 128(N); 8 waves; wave = 32x64 (2 M-tiles x 4 N-tiles).
// A/B tiles double-buffered in LDS via global_load_async_to_lds_b128.
// MODE 0: f32 out; MODE 1: bf16 out; MODE 2: exact GELU -> bf16 out;
// MODE 3: f32 accumulate C += rowScale[m]*(gemm+bias)
// ---------------------------------------------------------------------------
template <int MODE>
__global__ __launch_bounds__(256)
void gemm_bf16_wmma(const bf16_t* __restrict__ A, const bf16_t* __restrict__ W,
                    const float* __restrict__ bias,
                    float* __restrict__ Cf, bf16_t* __restrict__ Cb,
                    const float* __restrict__ rowScale, int rowScaleStride,
                    int M, int N, int K) {
  __shared__ bf16_t As[2][128][LDS_PITCH];
  __shared__ bf16_t Bs[2][128][LDS_PITCH];

  int lane = threadIdx.x & 31;
  int wave = threadIdx.x >> 5;
  int wm = wave & 3;                 // 0..3 : 32-row slab
  int wn = wave >> 2;                // 0..1 : 64-col slab
  int row0 = blockIdx.y * 128;
  int col0 = blockIdx.x * 128;

  v8f acc[2][4];
#pragma unroll
  for (int u = 0; u < 2; ++u)
#pragma unroll
    for (int t = 0; t < 4; ++t)
#pragma unroll
      for (int r = 0; r < 8; ++r) acc[u][t][r] = 0.0f;

  const bf16_t* Ab = A + (size_t)row0 * K;
  const bf16_t* Wb = W + (size_t)col0 * K;

  // prologue: stage K-step 0 into buffer 0
  stage_tile_async(Ab, K, As[0], 0);
  stage_tile_async(Wb, K, Bs[0], 0);

  int nk = K / 32;
  for (int kk = 0; kk < nk; ++kk) {
    int buf = kk & 1;
    wait_async0();
    __syncthreads();
    if (kk + 1 < nk) {
      stage_tile_async(Ab, K, As[buf ^ 1], (kk + 1) * 32);
      stage_tile_async(Wb, K, Bs[buf ^ 1], (kk + 1) * 32);
    }
    v16bf a0 = load_frag_A(&As[buf][wm * 32][0],      LDS_PITCH, lane);
    v16bf a1 = load_frag_A(&As[buf][wm * 32 + 16][0], LDS_PITCH, lane);
#pragma unroll
    for (int t = 0; t < 4; ++t) {
      v16bf bfr = load_frag_B_rows(&Bs[buf][wn * 64 + t * 16][0], LDS_PITCH, lane);
      acc[0][t] = wmma_bf16(a0, bfr, acc[0][t]);
      acc[1][t] = wmma_bf16(a1, bfr, acc[1][t]);
    }
    __syncthreads();
  }

  int nn   = lane & 15;
  int moff = (lane >> 4) << 3;
#pragma unroll
  for (int u = 0; u < 2; ++u) {
#pragma unroll
    for (int t = 0; t < 4; ++t) {
      int c = col0 + wn * 64 + t * 16 + nn;
      float bv = bias ? bias[c] : 0.0f;
#pragma unroll
      for (int r = 0; r < 8; ++r) {
        int m = row0 + wm * 32 + u * 16 + moff + r;
        float val = acc[u][t][r] + bv;
        size_t idx = (size_t)m * N + c;
        if (MODE == 0) {
          Cf[idx] = val;
        } else if (MODE == 1) {
          Cb[idx] = (bf16_t)val;
        } else if (MODE == 2) {
          float g = 0.5f * val * (1.0f + erff(val * 0.70710678118f));
          Cb[idx] = (bf16_t)g;
        } else {
          float s = rowScale[(size_t)m * rowScaleStride];
          Cf[idx] += s * val;
        }
      }
    }
  }
}

// ---------------------------------------------------------------------------
// Series decomposition: softmax(alpha)-mixed depthwise moving averages
// (k = 7/25/49, reflect padding).  One block per token (b,n); threads over D.
// ---------------------------------------------------------------------------
__global__ __launch_bounds__(256)
void decomp_kernel(const float* __restrict__ x, const float* __restrict__ alpha,
                   const float* __restrict__ dw7, const float* __restrict__ dw25,
                   const float* __restrict__ dw49,
                   float* __restrict__ Tr, float* __restrict__ S) {
  int bn = blockIdx.x;
  int b = bn >> 10;
  int n = bn & 1023;
  float a0 = alpha[0], a1 = alpha[1], a2 = alpha[2];
  float am = fmaxf(a0, fmaxf(a1, a2));
  float e0 = __expf(a0 - am), e1 = __expf(a1 - am), e2 = __expf(a2 - am);
  float inv = 1.0f / (e0 + e1 + e2);
  float w0 = e0 * inv, w1 = e1 * inv, w2 = e2 * inv;

  for (int d = threadIdx.x; d < D_MODEL; d += 256) {
    float acc = 0.0f;
    for (int j = 0; j < 49; ++j) {
      int p = n + j - 24;
      p = (p < 0) ? -p : ((p > 1023) ? 2046 - p : p);
      float coef = w2 * dw49[d * 49 + j];
      int j25 = j - 12;
      if (j25 >= 0 && j25 < 25) coef += w1 * dw25[d * 25 + j25];
      int j7 = j - 21;
      if (j7 >= 0 && j7 < 7) coef += w0 * dw7[d * 7 + j7];
      acc += coef * x[((size_t)b * SEQ_N + p) * D_MODEL + d];
    }
    size_t idx = (size_t)bn * D_MODEL + d;
    Tr[idx] = acc;
    S[idx]  = x[idx] - acc;
  }
}

// ---------------------------------------------------------------------------
// Block reduction helper (256 threads = 8 waves of 32)
// ---------------------------------------------------------------------------
__device__ inline float block_sum(float v, float* red) {
  int lane = threadIdx.x & 31;
  int wave = threadIdx.x >> 5;
#pragma unroll
  for (int m = 16; m; m >>= 1) v += __shfl_xor(v, m, 32);
  if (lane == 0) red[wave] = v;
  __syncthreads();
  float s = 0.0f;
#pragma unroll
  for (int i = 0; i < 8; ++i) s += red[i];
  __syncthreads();
  return s;
}

// LayerNorm over D; ADD=1 additionally computes x_s = X+Y and stores it.
template <int ADD>
__global__ __launch_bounds__(256)
void ln_kernel(const float* __restrict__ X, const float* __restrict__ Y,
               const float* __restrict__ g, const float* __restrict__ be,
               float* __restrict__ xs_out, bf16_t* __restrict__ outb) {
  __shared__ float red[8];
  size_t off = (size_t)blockIdx.x * D_MODEL;
  float vals[4];
  float s = 0.0f;
#pragma unroll
  for (int i = 0; i < 4; ++i) {
    int d = threadIdx.x + i * 256;
    float v = X[off + d];
    if (ADD) { v += Y[off + d]; xs_out[off + d] = v; }
    vals[i] = v;
    s += v;
  }
  float mean = block_sum(s, red) * (1.0f / 1024.0f);
  float vs = 0.0f;
#pragma unroll
  for (int i = 0; i < 4; ++i) { float dv = vals[i] - mean; vs += dv * dv; }
  float var = block_sum(vs, red) * (1.0f / 1024.0f);
  float rstd = rsqrtf(var + 1e-5f);
#pragma unroll
  for (int i = 0; i < 4; ++i) {
    int d = threadIdx.x + i * 256;
    outb[off + d] = (bf16_t)((vals[i] - mean) * rstd * g[d] + be[d]);
  }
}

// ---------------------------------------------------------------------------
// ALiBi attention, one block per (b,h, 32-query slab).
// Phase 1: logits = Q K^T via WMMA into 128 KB LDS f32 tile.
// Phase 2: ALiBi + softmax (register-cached rows), bf16 probs in-place, 1/sum saved.
// Phase 3: y = P V via WMMA (A from LDS), scaled store as bf16.
// Dynamic LDS: 32 rows * 4096 B + 128 B rowsum = 131200 B (of the 320 KB WGP pool).
// ---------------------------------------------------------------------------
__global__ __launch_bounds__(256)
void attn_kernel(const bf16_t* __restrict__ qb, const bf16_t* __restrict__ kb,
                 const bf16_t* __restrict__ vb, bf16_t* __restrict__ yb) {
  extern __shared__ char smem[];
  float*  lg = (float*)smem;                        // [32][1024] f32
  bf16_t* pball = (bf16_t*)smem;                    // in-place bf16, pitch 2048
  float*  rs = (float*)(smem + 32 * 4096);          // [32] 1/rowsum

  int lane = threadIdx.x & 31;
  int wave = threadIdx.x >> 5;
  int bh = blockIdx.y;
  int b = bh >> 4;
  int h = bh & 15;
  int q0 = blockIdx.x * 32;
  size_t base = ((size_t)b * SEQ_N) * D_MODEL + (size_t)h * HEAD_DIM;

  // ---- Phase 1: logits ----
  v16bf a00 = load_frag_A(qb + base + (size_t)q0 * D_MODEL + 0,  D_MODEL, lane);
  v16bf a01 = load_frag_A(qb + base + (size_t)q0 * D_MODEL + 32, D_MODEL, lane);
  v16bf a10 = load_frag_A(qb + base + (size_t)(q0 + 16) * D_MODEL + 0,  D_MODEL, lane);
  v16bf a11 = load_frag_A(qb + base + (size_t)(q0 + 16) * D_MODEL + 32, D_MODEL, lane);

  int cbase = wave * 128;
  int nn   = lane & 15;
  int moff = (lane >> 4) << 3;
#pragma unroll 1
  for (int t = 0; t < 8; ++t) {
    int n0 = cbase + t * 16;
    v16bf b0 = load_frag_B_rows(kb + base + (size_t)n0 * D_MODEL + 0,  D_MODEL, lane);
    v16bf b1 = load_frag_B_rows(kb + base + (size_t)n0 * D_MODEL + 32, D_MODEL, lane);
    v8f acc0, acc1;
#pragma unroll
    for (int r = 0; r < 8; ++r) { acc0[r] = 0.0f; acc1[r] = 0.0f; }
    acc0 = wmma_bf16(a00, b0, acc0);
    acc0 = wmma_bf16(a01, b1, acc0);
    acc1 = wmma_bf16(a10, b0, acc1);
    acc1 = wmma_bf16(a11, b1, acc1);
#pragma unroll
    for (int r = 0; r < 8; ++r) {
      lg[(moff + r) * 1024 + n0 + nn]        = acc0[r];
      lg[(16 + moff + r) * 1024 + n0 + nn]   = acc1[r];
    }
  }
  __syncthreads();

  // ---- Phase 2: ALiBi softmax ----
  float slope = exp2f(-0.5f * (float)(h + 1));
  const float scale = 0.125f;  // 1/sqrt(64)
  for (int r = wave * 4; r < wave * 4 + 4; ++r) {
    int qi = q0 + r;
    float* lr = lg + (size_t)r * 1024;
    bf16_t* pr = pball + (size_t)r * 2048;
    float vals[32];
    float mx = -1e30f;
#pragma unroll
    for (int i = 0; i < 32; ++i) {
      int c = lane + i * 32;
      float pen = (c > qi) ? slope * (float)(c - qi) : 0.0f;
      float v = lr[c] * scale - pen;
      vals[i] = v;
      mx = fmaxf(mx, v);
    }
#pragma unroll
    for (int m = 16; m; m >>= 1) mx = fmaxf(mx, __shfl_xor(mx, m, 32));
    float sum = 0.0f;
#pragma unroll
    for (int i = 0; i < 32; ++i) {
      float e = __expf(vals[i] - mx);
      sum += e;
      pr[lane + i * 32] = (bf16_t)e;
    }
#pragma unroll
    for (int m = 16; m; m >>= 1) sum += __shfl_xor(sum, m, 32);
    if (lane == 0) rs[r] = 1.0f / sum;
  }
  __syncthreads();

  // ---- Phase 3: y = P V ----
  int mt = wave >> 2;   // 0..1
  int nt = wave & 3;    // 0..3
  v8f acc;
#pragma unroll
  for (int r = 0; r < 8; ++r) acc[r] = 0.0f;
  const bf16_t* Abase = pball + (size_t)(mt * 16) * 2048;
  for (int k0 = 0; k0 < SEQ_N; k0 += 32) {
    v16bf af = load_frag_A(Abase + k0, 2048, lane);
    v16bf bf_ = load_frag_B_cols(vb + base + (size_t)k0 * D_MODEL + nt * 16, D_MODEL, lane);
    acc = wmma_bf16(af, bf_, acc);
  }
#pragma unroll
  for (int r = 0; r < 8; ++r) {
    int m = mt * 16 + moff + r;
    float val = acc[r] * rs[m];
    yb[base + (size_t)(q0 + m) * D_MODEL + nt * 16 + nn] = (bf16_t)val;
  }
}

// ---------------------------------------------------------------------------
// Router: gates = softmax(xf @ rw^T) (E=4), top-2 normalized weights.
// ---------------------------------------------------------------------------
__global__ __launch_bounds__(256)
void router_kernel(const bf16_t* __restrict__ xf, const float* __restrict__ rw,
                   float* __restrict__ gates, float* __restrict__ wts) {
  __shared__ float red[8];
  __shared__ float gsh[4];
  size_t off = (size_t)blockIdx.x * D_MODEL;
  float acc[4] = {0.f, 0.f, 0.f, 0.f};
#pragma unroll
  for (int i = 0; i < 4; ++i) {
    int d = threadIdx.x + i * 256;
    float xv = (float)xf[off + d];
#pragma unroll
    for (int e = 0; e < 4; ++e) acc[e] += xv * rw[e * D_MODEL + d];
  }
#pragma unroll 1
  for (int e = 0; e < 4; ++e) {
    float s = block_sum(acc[e], red);
    if (threadIdx.x == 0) gsh[e] = s;
  }
  __syncthreads();
  if (threadIdx.x == 0) {
    float mx = fmaxf(fmaxf(gsh[0], gsh[1]), fmaxf(gsh[2], gsh[3]));
    float ex[4], se = 0.0f;
#pragma unroll
    for (int e = 0; e < 4; ++e) { ex[e] = __expf(gsh[e] - mx); se += ex[e]; }
    float gg[4];
#pragma unroll
    for (int e = 0; e < 4; ++e) gg[e] = ex[e] / se;
    int i1 = 0;
    for (int j = 1; j < 4; ++j) if (gg[j] > gg[i1]) i1 = j;
    int i2 = (i1 == 0) ? 1 : 0;
    for (int j = 0; j < 4; ++j) if (j != i1 && gg[j] > gg[i2]) i2 = j;
    float s2 = fmaxf(gg[i1] + gg[i2], 1e-9f);
#pragma unroll
    for (int e = 0; e < 4; ++e) {
      gates[(size_t)blockIdx.x * 4 + e] = gg[e];
      wts[(size_t)blockIdx.x * 4 + e] =
          (e == i1) ? gg[i1] / s2 : ((e == i2) ? gg[i2] / s2 : 0.0f);
    }
  }
}

// Deterministic aux-loss reduction, single block.
__global__ __launch_bounds__(256)
void aux_kernel(const float* __restrict__ gates, const float* __restrict__ wts,
                float* __restrict__ aux_out) {
  __shared__ float red[8];
  float ag[4] = {0.f, 0.f, 0.f, 0.f};
  float as[4] = {0.f, 0.f, 0.f, 0.f};
  for (int t = threadIdx.x; t < TOKENS; t += 256) {
#pragma unroll
    for (int e = 0; e < 4; ++e) {
      ag[e] += gates[(size_t)t * 4 + e];
      as[e] += (wts[(size_t)t * 4 + e] > 0.0f) ? 1.0f : 0.0f;
    }
  }
  float aux = 0.0f;
#pragma unroll 1
  for (int e = 0; e < 4; ++e) {
    float A  = block_sum(ag[e], red);
    float Bv = block_sum(as[e], red);
    aux += (A * (1.0f / TOKENS)) * (Bv * (1.0f / TOKENS));
  }
  if (threadIdx.x == 0) aux_out[0] = 4.0f * aux;
}

// Final: out = x_s + moe + dwconv5(Tr, zero-pad 2) + tb
__global__ __launch_bounds__(256)
void final_kernel(const float* __restrict__ xs, const float* __restrict__ moe,
                  const float* __restrict__ Tr, const float* __restrict__ tw,
                  const float* __restrict__ tb_, float* __restrict__ out) {
  int bn = blockIdx.x;
  int b = bn >> 10;
  int n = bn & 1023;
  for (int d = threadIdx.x; d < D_MODEL; d += 256) {
    float t = tb_[d];
#pragma unroll
    for (int i = 0; i < 5; ++i) {
      int p = n + i - 2;
      if (p >= 0 && p < SEQ_N)
        t += tw[d * 5 + i] * Tr[((size_t)b * SEQ_N + p) * D_MODEL + d];
    }
    size_t idx = (size_t)bn * D_MODEL + d;
    out[idx] = xs[idx] + moe[idx] + t;
  }
}

__global__ void zero_kernel(float* p, int n) {
  int i = blockIdx.x * blockDim.x + threadIdx.x;
  if (i < n) p[i] = 0.0f;
}

__global__ void cvt_kernel(const float* __restrict__ s, bf16_t* __restrict__ d, int n) {
  int i = blockIdx.x * blockDim.x + threadIdx.x;
  if (i < n) d[i] = (bf16_t)s[i];
}

// ---------------------------------------------------------------------------
// Host-side orchestration
// ---------------------------------------------------------------------------
extern "C" void kernel_launch(void* const* d_in, const int* in_sizes, int n_in,
                              void* d_out, int out_size, void* d_ws, size_t ws_size,
                              hipStream_t stream) {
  (void)in_sizes; (void)n_in; (void)out_size; (void)ws_size;
  const float* x    = (const float*)d_in[0];
  const float* qw   = (const float*)d_in[1];  const float* qbias = (const float*)d_in[2];
  const float* kw   = (const float*)d_in[3];  const float* kbias = (const float*)d_in[4];
  const float* vw   = (const float*)d_in[5];  const float* vbias = (const float*)d_in[6];
  const float* ow   = (const float*)d_in[7];  const float* obias = (const float*)d_in[8];
  const float* n1g  = (const float*)d_in[9];  const float* n1b   = (const float*)d_in[10];
  const float* n2g  = (const float*)d_in[11]; const float* n2b   = (const float*)d_in[12];
  const float* alpha= (const float*)d_in[13];
  const float* dw7  = (const float*)d_in[14];
  const float* dw25 = (const float*)d_in[15];
  const float* dw49 = (const float*)d_in[16];
  const float* rw   = (const float*)d_in[17];
  const float* ew1  = (const float*)d_in[18]; const float* eb1 = (const float*)d_in[19];
  const float* ew2  = (const float*)d_in[20]; const float* eb2 = (const float*)d_in[21];
  const float* tw   = (const float*)d_in[22]; const float* tb  = (const float*)d_in[23];
  float* out = (float*)d_out;

  const size_t ND = (size_t)TOKENS * D_MODEL;  // 8388608

  // ---- workspace carve ----
  char* w = (char*)d_ws;
  auto alloc = [&](size_t bytes) {
    char* p = w;
    w += (bytes + 255) & ~(size_t)255;
    return p;
  };
  float*  Tr    = (float*)alloc(ND * 4);
  float*  S     = (float*)alloc(ND * 4);
  float*  yproj = (float*)alloc(ND * 4);
  float*  xs    = (float*)alloc(ND * 4);
  float*  moe   = (float*)alloc(ND * 4);
  bf16_t* Snb   = (bf16_t*)alloc(ND * 2);
  bf16_t* qB    = (bf16_t*)alloc(ND * 2);
  bf16_t* kB    = (bf16_t*)alloc(ND * 2);
  bf16_t* vB    = (bf16_t*)alloc(ND * 2);
  bf16_t* yB    = (bf16_t*)alloc(ND * 2);
  bf16_t* xfB   = (bf16_t*)alloc(ND * 2);
  bf16_t* hB    = (bf16_t*)alloc((size_t)TOKENS * MOE_HID * 2);
  bf16_t* qwB   = (bf16_t*)alloc((size_t)D_MODEL * D_MODEL * 2);
  bf16_t* kwB   = (bf16_t*)alloc((size_t)D_MODEL * D_MODEL * 2);
  bf16_t* vwB   = (bf16_t*)alloc((size_t)D_MODEL * D_MODEL * 2);
  bf16_t* owB   = (bf16_t*)alloc((size_t)D_MODEL * D_MODEL * 2);
  bf16_t* ew1B  = (bf16_t*)alloc((size_t)NEXP * MOE_HID * D_MODEL * 2);
  bf16_t* ew2B  = (bf16_t*)alloc((size_t)NEXP * D_MODEL * MOE_HID * 2);
  float*  wts   = (float*)alloc((size_t)TOKENS * 4 * 4);
  float*  gates = (float*)alloc((size_t)TOKENS * 4 * 4);

  const int CVT_T = 256;
  auto cvt = [&](const float* src, bf16_t* dst, size_t n) {
    cvt_kernel<<<dim3((unsigned)((n + CVT_T - 1) / CVT_T)), CVT_T, 0, stream>>>(src, dst, (int)n);
  };

  // 1) weight conversions + zero accumulator
  cvt(qw, qwB, (size_t)D_MODEL * D_MODEL);
  cvt(kw, kwB, (size_t)D_MODEL * D_MODEL);
  cvt(vw, vwB, (size_t)D_MODEL * D_MODEL);
  cvt(ow, owB, (size_t)D_MODEL * D_MODEL);
  cvt(ew1, ew1B, (size_t)NEXP * MOE_HID * D_MODEL);
  cvt(ew2, ew2B, (size_t)NEXP * D_MODEL * MOE_HID);
  zero_kernel<<<dim3((unsigned)((ND + 255) / 256)), 256, 0, stream>>>(moe, (int)ND);

  // 2) decomposition
  decomp_kernel<<<dim3(TOKENS), 256, 0, stream>>>(x, alpha, dw7, dw25, dw49, Tr, S);

  // 3) LN1 -> Sn (bf16)
  ln_kernel<0><<<dim3(TOKENS), 256, 0, stream>>>(S, nullptr, n1g, n1b, nullptr, Snb);

  // 4) QKV projections (bf16 out); 128x128 block tiles
  dim3 gemm_grid_d(D_MODEL / 128, TOKENS / 128);
  gemm_bf16_wmma<1><<<gemm_grid_d, 256, 0, stream>>>(Snb, qwB, qbias, nullptr, qB,
                                                     nullptr, 0, TOKENS, D_MODEL, D_MODEL);
  gemm_bf16_wmma<1><<<gemm_grid_d, 256, 0, stream>>>(Snb, kwB, kbias, nullptr, kB,
                                                     nullptr, 0, TOKENS, D_MODEL, D_MODEL);
  gemm_bf16_wmma<1><<<gemm_grid_d, 256, 0, stream>>>(Snb, vwB, vbias, nullptr, vB,
                                                     nullptr, 0, TOKENS, D_MODEL, D_MODEL);

  // 5) ALiBi attention (dynamic LDS 131200 B out of the 320 KB WGP pool)
  const int ATTN_SMEM = 32 * 4096 + 128;
  hipFuncSetAttribute((const void*)attn_kernel,
                      hipFuncAttributeMaxDynamicSharedMemorySize, ATTN_SMEM);
  attn_kernel<<<dim3(SEQ_N / 32, BATCH * NHEAD), 256, ATTN_SMEM, stream>>>(qB, kB, vB, yB);

  // 6) output projection (f32 out) + LN2 (residual S + y, emit x_s and xf)
  gemm_bf16_wmma<0><<<gemm_grid_d, 256, 0, stream>>>(yB, owB, obias, yproj, nullptr,
                                                     nullptr, 0, TOKENS, D_MODEL, D_MODEL);
  ln_kernel<1><<<dim3(TOKENS), 256, 0, stream>>>(S, yproj, n2g, n2b, xs, xfB);

  // 7) router + aux
  router_kernel<<<dim3(TOKENS), 256, 0, stream>>>(xfB, rw, gates, wts);
  aux_kernel<<<dim3(1), 256, 0, stream>>>(gates, wts, out + ND);

  // 8) MoE experts (dense evaluation, gate-scaled accumulation)
  dim3 gemm_grid_h(MOE_HID / 128, TOKENS / 128);
  for (int e = 0; e < NEXP; ++e) {
    const bf16_t* w1 = ew1B + (size_t)e * MOE_HID * D_MODEL;
    const bf16_t* w2 = ew2B + (size_t)e * D_MODEL * MOE_HID;
    gemm_bf16_wmma<2><<<gemm_grid_h, 256, 0, stream>>>(
        xfB, w1, eb1 + (size_t)e * MOE_HID, nullptr, hB, nullptr, 0,
        TOKENS, MOE_HID, D_MODEL);
    gemm_bf16_wmma<3><<<gemm_grid_d, 256, 0, stream>>>(
        hB, w2, eb2 + (size_t)e * D_MODEL, moe, nullptr, wts + e, 4,
        TOKENS, D_MODEL, MOE_HID);
  }

  // 9) final fusion: x_s + moe + trend conv
  final_kernel<<<dim3(TOKENS), 256, 0, stream>>>(xs, moe, Tr, tw, tb, out);
}
